// FlowNet3D_51247549776068
// MI455X (gfx1250) — compile-verified, matrix operand order checked by reading, not run on hardware
//
#include <hip/hip_runtime.h>

typedef __attribute__((ext_vector_type(16))) _Float16 v16h;
typedef __attribute__((ext_vector_type(8)))  _Float16 v8h;
typedef __attribute__((ext_vector_type(4)))  _Float16 v4h;
typedef __attribute__((ext_vector_type(8)))  float    v8f;

#define NEG_HUGE (-3.0e38f)

// ---------------------------------------------------------------------------
// features [B,C,N] -> [B,N,C]
__global__ void k_transpose(const float* __restrict__ src, float* __restrict__ dst,
                            int Bv, int Cch, int Np) {
  int t = blockIdx.x * blockDim.x + threadIdx.x;
  int total = Bv * Cch * Np;
  if (t >= total) return;
  int n = t % Np;
  int c = (t / Np) % Cch;
  int b = t / (Np * Cch);
  dst[((size_t)b * Np + n) * Cch + c] = src[t];
}

// ---------------------------------------------------------------------------
// farthest point sampling; one block per batch, 256 threads
__global__ void k_fps(const float* __restrict__ xyz, int Np, int npoint,
                      int* __restrict__ idx_out, float* __restrict__ new_xyz) {
  __shared__ float dist[4096];
  __shared__ float smax[256];
  __shared__ int   simax[256];
  __shared__ int   far_s;
  int b = blockIdx.x;
  int tid = threadIdx.x;
  const float* p = xyz + (size_t)b * Np * 3;
  for (int i = tid; i < Np; i += 256) dist[i] = 1e10f;
  if (tid == 0) far_s = 0;
  __syncthreads();
  for (int it = 0; it < npoint; ++it) {
    int far = far_s;
    if (tid == 0) {
      idx_out[b * npoint + it] = far;
      new_xyz[((size_t)b * npoint + it) * 3 + 0] = p[far * 3 + 0];
      new_xyz[((size_t)b * npoint + it) * 3 + 1] = p[far * 3 + 1];
      new_xyz[((size_t)b * npoint + it) * 3 + 2] = p[far * 3 + 2];
    }
    float cx = p[far * 3 + 0], cy = p[far * 3 + 1], cz = p[far * 3 + 2];
    float best = NEG_HUGE; int besti = 0;
    for (int i = tid; i < Np; i += 256) {
      float dx = p[i * 3 + 0] - cx, dy = p[i * 3 + 1] - cy, dz = p[i * 3 + 2] - cz;
      float d = dx * dx + dy * dy + dz * dz;
      float nd = fminf(dist[i], d);
      dist[i] = nd;
      if (nd > best) { best = nd; besti = i; }
    }
    smax[tid] = best; simax[tid] = besti;
    __syncthreads();
    for (int s = 128; s > 0; s >>= 1) {
      if (tid < s && smax[tid + s] > smax[tid]) { smax[tid] = smax[tid + s]; simax[tid] = simax[tid + s]; }
      __syncthreads();
    }
    if (tid == 0) far_s = simax[0];
    __syncthreads();
  }
}

// ---------------------------------------------------------------------------
// ball query + gather + concat([feat, rel_xyz]) with zero pad up to KP
// one block per (b,s)
__global__ void k_group(const float* __restrict__ pts, const float* __restrict__ feat,
                        const float* __restrict__ new_xyz,
                        int Np, int C, int S, int K, int KP, float radius2,
                        float* __restrict__ X) {
  __shared__ int nidx[16];
  int bs = blockIdx.x;
  int b  = bs / S;
  const float* P = pts + (size_t)b * Np * 3;
  float qx = new_xyz[(size_t)bs * 3 + 0];
  float qy = new_xyz[(size_t)bs * 3 + 1];
  float qz = new_xyz[(size_t)bs * 3 + 2];
  if (threadIdx.x == 0) {
    int cnt = 0;
    for (int i = 0; i < Np && cnt < K; ++i) {
      float dx = P[i * 3 + 0] - qx, dy = P[i * 3 + 1] - qy, dz = P[i * 3 + 2] - qz;
      if (dx * dx + dy * dy + dz * dz <= radius2) nidx[cnt++] = i;
    }
    if (cnt == 0) nidx[cnt++] = 0;
    for (int k = cnt; k < K; ++k) nidx[k] = nidx[0];
  }
  __syncthreads();
  const float* F = feat + (size_t)b * Np * C;
  float* Xr = X + (size_t)bs * K * KP;
  int total = K * KP;
  for (int t = threadIdx.x; t < total; t += blockDim.x) {
    int k = t / KP, c = t % KP;
    int id = nidx[k];
    float v = 0.0f;
    if (c < C) v = F[(size_t)id * C + c];
    else if (c < C + 3) {
      int a = c - C;
      float q = (a == 0) ? qx : ((a == 1) ? qy : qz);
      v = P[id * 3 + a] - q;
    }
    Xr[t] = v;
  }
}

// ---------------------------------------------------------------------------
// generic kNN: per-query repeated-min with lexicographic exclusion
__global__ void k_knn(const float* __restrict__ query, const float* __restrict__ ref,
                      int Bv, int Q, int M, int K,
                      int* __restrict__ idx_out, float* __restrict__ d_out) {
  int t = blockIdx.x * blockDim.x + threadIdx.x;
  if (t >= Bv * Q) return;
  int b = t / Q;
  const float* qp = query + (size_t)t * 3;
  const float* rp = ref + (size_t)b * M * 3;
  float qx = qp[0], qy = qp[1], qz = qp[2];
  float lastd = NEG_HUGE; int lasti = -1;
  for (int k = 0; k < K; ++k) {
    float bd = 3.0e38f; int bi = -1;
    for (int j = 0; j < M; ++j) {
      float dx = rp[j * 3 + 0] - qx, dy = rp[j * 3 + 1] - qy, dz = rp[j * 3 + 2] - qz;
      float d = dx * dx + dy * dy + dz * dz;
      bool after = (d > lastd) || (d == lastd && j > lasti);
      if (after && ((bi < 0) || (d < bd))) { bd = d; bi = j; }
    }
    idx_out[(size_t)t * K + k] = bi;
    if (d_out) d_out[(size_t)t * K + k] = bd;
    lastd = bd; lasti = bi;
  }
}

// ---------------------------------------------------------------------------
// flow-embedding grouped input for ONE batch: concat([f2(128), f1(128), dxyz(3)])
__global__ void k_group_fe(const float* __restrict__ f1b, const float* __restrict__ f2b,
                           const float* __restrict__ p1b, const float* __restrict__ p2b,
                           const int* __restrict__ fib, int S, int Kn, int KP,
                           float* __restrict__ X) {
  int t = blockIdx.x * blockDim.x + threadIdx.x;
  int total = S * Kn * KP;
  if (t >= total) return;
  int c = t % KP;
  int sk = t / KP;
  int k = sk % Kn, s = sk / Kn;
  int id = fib[s * Kn + k];
  float v = 0.0f;
  if (c < 128) v = f2b[(size_t)id * 128 + c];
  else if (c < 256) v = f1b[(size_t)s * 128 + (c - 128)];
  else if (c < 259) { int a = c - 256; v = p2b[id * 3 + a] - p1b[s * 3 + a]; }
  X[t] = v;
}

// ---------------------------------------------------------------------------
// up1: x = max_k concat([f1_4[ui], p1_4[ui]-p1_3]) then concat f1_3, pad->800
__global__ void k_up1_build(const float* __restrict__ f1_4, const float* __restrict__ p1_4,
                            const float* __restrict__ p1_3, const float* __restrict__ f1_3,
                            const int* __restrict__ ui, int Bv, float* __restrict__ X) {
  int t = blockIdx.x * blockDim.x + threadIdx.x;
  int total = Bv * 64 * 800;
  if (t >= total) return;
  int c = t % 800;
  int bd = t / 800;
  int b = bd / 64;
  const int* u = ui + (size_t)bd * 8;
  float v = 0.0f;
  if (c < 512) {
    v = NEG_HUGE;
    for (int k = 0; k < 8; ++k) v = fmaxf(v, f1_4[((size_t)b * 16 + u[k]) * 512 + c]);
  } else if (c < 515) {
    int a = c - 512;
    v = NEG_HUGE;
    for (int k = 0; k < 8; ++k)
      v = fmaxf(v, p1_4[((size_t)b * 16 + u[k]) * 3 + a] - p1_3[(size_t)bd * 3 + a]);
  } else if (c < 771) {
    v = f1_3[(size_t)bd * 256 + (c - 515)];
  }
  X[t] = v;
}

// ---------------------------------------------------------------------------
// up2 grouped input: concat([nf1_3[ui](256), dxyz(3)]) pad->288
__global__ void k_group_up2(const float* __restrict__ fs, const float* __restrict__ ps,
                            const float* __restrict__ pd, const int* __restrict__ ui,
                            int Bv, float* __restrict__ X) {
  int t = blockIdx.x * blockDim.x + threadIdx.x;
  int total = Bv * 256 * 8 * 288;
  if (t >= total) return;
  int c = t % 288;
  int r = t / 288;            // (b*256+s)*8 + k
  int s2 = r / 8;             // b*256+s
  int b = s2 / 256;
  int id = ui[r];
  float v = 0.0f;
  if (c < 256) v = fs[((size_t)b * 64 + id) * 256 + c];
  else if (c < 259) { int a = c - 256; v = ps[((size_t)b * 64 + id) * 3 + a] - pd[(size_t)s2 * 3 + a]; }
  X[t] = v;
}

// ---------------------------------------------------------------------------
// concat [maxed(256), f1_2(128), emb(128)] -> 512
__global__ void k_cat_up2(const float* __restrict__ xm, const float* __restrict__ f1_2,
                          const float* __restrict__ emb, int Bv, float* __restrict__ X) {
  int t = blockIdx.x * blockDim.x + threadIdx.x;
  int total = Bv * 256 * 512;
  if (t >= total) return;
  int c = t % 512;
  int r = t / 512;
  float v;
  if (c < 256)      v = xm[(size_t)r * 256 + c];
  else if (c < 384) v = f1_2[(size_t)r * 128 + (c - 256)];
  else              v = emb[(size_t)r * 128 + (c - 384)];
  X[t] = v;
}

// ---------------------------------------------------------------------------
// feature propagation: 3-NN inverse-sqdist interp of nf1_2 + concat f1
__global__ void k_fp_build(const float* __restrict__ nf12, const float* __restrict__ f1t,
                           const int* __restrict__ pi, const float* __restrict__ pd2,
                           int Bv, int Nv, float* __restrict__ X) {
  int t = blockIdx.x * blockDim.x + threadIdx.x;
  int total = Bv * Nv * 320;
  if (t >= total) return;
  int c = t % 320;
  int bn = t / 320;
  int b = bn / Nv;
  if (c < 256) {
    const int* ii = pi + (size_t)bn * 3;
    const float* dd = pd2 + (size_t)bn * 3;
    float w0 = 1.0f / fmaxf(dd[0], 1e-10f);
    float w1 = 1.0f / fmaxf(dd[1], 1e-10f);
    float w2 = 1.0f / fmaxf(dd[2], 1e-10f);
    float ws = w0 + w1 + w2;
    const float* F = nf12 + (size_t)b * 256 * 256;
    X[t] = (w0 * F[(size_t)ii[0] * 256 + c] + w1 * F[(size_t)ii[1] * 256 + c] +
            w2 * F[(size_t)ii[2] * 256 + c]) / ws;
  } else {
    X[t] = f1t[(size_t)bn * 64 + (c - 256)];
  }
}

// ---------------------------------------------------------------------------
// max over neighbor axis: X [Srows,K,C] -> Y [Srows,C]
__global__ void k_maxpool(const float* __restrict__ X, float* __restrict__ Y,
                          int Srows, int K, int C) {
  int t = blockIdx.x * blockDim.x + threadIdx.x;
  if (t >= Srows * C) return;
  int s = t / C, c = t % C;
  const float* p = X + ((size_t)s * K) * C + c;
  float m = p[0];
  for (int k = 1; k < K; ++k) m = fmaxf(m, p[(size_t)k * C]);
  Y[t] = m;
}

// ---------------------------------------------------------------------------
// fold eval-mode BN into f16 weights + f32 bias. W [Cout,Cin] -> W16 [Cout,KP]
__global__ void k_prep_w(const float* __restrict__ W, const float* __restrict__ bvec,
                         const float* __restrict__ be, const float* __restrict__ g,
                         const float* __restrict__ mu, const float* __restrict__ var,
                         int Cin, int KP, int Cout,
                         _Float16* __restrict__ W16, float* __restrict__ bias) {
  int t = blockIdx.x * blockDim.x + threadIdx.x;
  if (t >= Cout * KP) return;
  int o = t / KP, c = t % KP;
  float s = g[o] * rsqrtf(var[o] + 1e-3f);
  W16[t] = (c < Cin) ? (_Float16)(W[(size_t)o * Cin + c] * s) : (_Float16)0.0f;
  if (c == 0) bias[o] = s * (bvec[o] - mu[o]) + be[o];
}

// ---------------------------------------------------------------------------
// WMMA GEMM: Y[R,Cout] = relu(X[R,KP] * W16[Cout,KP]^T + bias)
// block = 8 waves; block tile 128(M) x 64(N); each wave owns a 16x64 strip
// and issues 4 back-to-back v_wmma_f32_16x16x32_f16 per K-chunk (A reused 4x).
__global__ __launch_bounds__(256) void k_gemm(
    const float* __restrict__ X, const _Float16* __restrict__ W16,
    const float* __restrict__ bias, float* __restrict__ Y,
    int R, int KP, int Cout, int relu) {
  __shared__ _Float16 As[128][32];   // 8 KB
  __shared__ _Float16 Bs[64][32];    // 4 KB
  const int tid  = threadIdx.x;
  const int wave = tid >> 5;
  const int lane = tid & 31;
  const int half = lane >> 4;
  const int l16  = lane & 15;
  const int row0 = blockIdx.x * 128;
  const int col0 = blockIdx.y * 64;

  v8f acc[4] = {{}, {}, {}, {}};

  const int am = tid >> 1;          // A staging: row 0..127
  const int ak = (tid & 1) * 16;    // 16 halves each
  const int bn_ = tid >> 2;         // B staging: col 0..63
  const int bk = (tid & 3) * 8;     // 8 halves each

  for (int k0 = 0; k0 < KP; k0 += 32) {
    {
      int gr = row0 + am;
      v8h h0 = {}, h1 = {};
      if (gr < R) {
        const float* s = X + (size_t)gr * KP + k0 + ak;
        const float4 f0 = *reinterpret_cast<const float4*>(s + 0);
        const float4 f1 = *reinterpret_cast<const float4*>(s + 4);
        const float4 f2 = *reinterpret_cast<const float4*>(s + 8);
        const float4 f3 = *reinterpret_cast<const float4*>(s + 12);
        h0[0] = (_Float16)f0.x; h0[1] = (_Float16)f0.y; h0[2] = (_Float16)f0.z; h0[3] = (_Float16)f0.w;
        h0[4] = (_Float16)f1.x; h0[5] = (_Float16)f1.y; h0[6] = (_Float16)f1.z; h0[7] = (_Float16)f1.w;
        h1[0] = (_Float16)f2.x; h1[1] = (_Float16)f2.y; h1[2] = (_Float16)f2.z; h1[3] = (_Float16)f2.w;
        h1[4] = (_Float16)f3.x; h1[5] = (_Float16)f3.y; h1[6] = (_Float16)f3.z; h1[7] = (_Float16)f3.w;
      }
      *reinterpret_cast<v8h*>(&As[am][ak + 0]) = h0;
      *reinterpret_cast<v8h*>(&As[am][ak + 8]) = h1;
    }
    {
      int gc = col0 + bn_;
      v8h h = {};
      if (gc < Cout) h = *reinterpret_cast<const v8h*>(W16 + (size_t)gc * KP + k0 + bk);
      *reinterpret_cast<v8h*>(&Bs[bn_][bk]) = h;
    }
    // prefetch next K-chunk (global_prefetch_b8 on gfx1250)
    if (k0 + 32 < KP) {
      int gr = row0 + am;
      if (gr < R) __builtin_prefetch(X + (size_t)gr * KP + k0 + 32 + ak, 0, 1);
      int gc = col0 + bn_;
      if (gc < Cout) __builtin_prefetch(W16 + (size_t)gc * KP + k0 + 32 + bk, 0, 1);
    }
    __syncthreads();
    // fragment build per CDNA5 16-bit A/B layout:
    // lanes 0-15 hold K base..base+7 and 16+base.., base = half*8
    const int base = half * 8;
    union HU { v16h v; v8h h[2]; };
    HU a;
    a.h[0] = *reinterpret_cast<const v8h*>(&As[wave * 16 + l16][base]);
    a.h[1] = *reinterpret_cast<const v8h*>(&As[wave * 16 + l16][16 + base]);
#pragma unroll
    for (int j = 0; j < 4; ++j) {
      HU bf;
      bf.h[0] = *reinterpret_cast<const v8h*>(&Bs[j * 16 + l16][base]);
      bf.h[1] = *reinterpret_cast<const v8h*>(&Bs[j * 16 + l16][16 + base]);
      acc[j] = __builtin_amdgcn_wmma_f32_16x16x32_f16(false, a.v, false, bf.v,
                                                      (short)0, acc[j], false, false);
    }
    __syncthreads();
  }

#pragma unroll
  for (int j = 0; j < 4; ++j) {
#pragma unroll
    for (int r = 0; r < 8; ++r) {
      int m = row0 + wave * 16 + r + half * 8;
      int n = col0 + j * 16 + l16;
      if (m < R && n < Cout) {
        float v = acc[j][r] + bias[n];
        if (relu) v = fmaxf(v, 0.0f);
        Y[(size_t)m * Cout + n] = v;
      }
    }
  }
}

// ---------------------------------------------------------------------------
// final classifier conv 128->3 (no BN), output transposed to [B,3,N]
__global__ void k_cls2(const float* __restrict__ X, const float* __restrict__ W,
                       const float* __restrict__ b3, int Bv, int Nv,
                       float* __restrict__ out) {
  int t = blockIdx.x * blockDim.x + threadIdx.x;
  if (t >= Bv * Nv) return;
  int b = t / Nv, n = t % Nv;
  const float* x = X + (size_t)t * 128;
  for (int o = 0; o < 3; ++o) {
    float s = b3[o];
    for (int c = 0; c < 128; ++c) s += x[c] * W[o * 128 + c];
    out[((size_t)b * 3 + o) * Nv + n] = s;
  }
}

// ===========================================================================
extern "C" void kernel_launch(void* const* d_in, const int* in_sizes, int n_in,
                              void* d_out, int out_size, void* d_ws, size_t ws_size,
                              hipStream_t stream) {
  (void)in_sizes; (void)n_in; (void)out_size; (void)ws_size;
  const int B = 16, N = 4096;

  const float* points1   = (const float*)d_in[0];
  const float* points2   = (const float*)d_in[1];
  const float* features1 = (const float*)d_in[2];
  const float* features2 = (const float*)d_in[3];
  auto pl = [&](int i) -> const float* { return (const float*)d_in[4 + i]; };

  // ---- workspace carve-up ----
  size_t off = 0;
  char* base = (char*)d_ws;
  auto allocB = [&](size_t bytes) -> void* {
    off = (off + 255) & ~(size_t)255;
    void* p = base + off; off += bytes; return p;
  };
  auto allocF = [&](size_t n) -> float*    { return (float*)allocB(n * 4); };
  auto allocI = [&](size_t n) -> int*      { return (int*)allocB(n * 4); };
  auto allocH = [&](size_t n) -> _Float16* { return (_Float16*)allocB(n * 2); };

  float* f1t  = allocF((size_t)B * N * 64);
  float* f2t  = allocF((size_t)B * N * 64);
  float* p1_2 = allocF((size_t)B * 256 * 3);
  float* p2_2 = allocF((size_t)B * 256 * 3);
  float* f1_2 = allocF((size_t)B * 256 * 128);
  float* f2_2 = allocF((size_t)B * 256 * 128);
  float* emb  = allocF((size_t)B * 256 * 128);
  float* p1_3 = allocF((size_t)B * 64 * 3);
  float* f1_3 = allocF((size_t)B * 64 * 256);
  float* p1_4 = allocF((size_t)B * 16 * 3);
  float* f1_4 = allocF((size_t)B * 16 * 512);
  float* nf13 = allocF((size_t)B * 64 * 256);
  float* nf12 = allocF((size_t)B * 256 * 256);
  int*   fpsi = allocI((size_t)B * 256);
  int*   fi   = allocI((size_t)B * 256 * 64);
  int*   ui1  = allocI((size_t)B * 64 * 8);
  int*   ui2  = allocI((size_t)B * 256 * 8);
  int*   pi   = allocI((size_t)B * N * 3);
  float* pd2  = allocF((size_t)B * N * 3);
  float* XA   = allocF((size_t)B * N * 320);  // big staging (84 MB)
  float* XB   = allocF((size_t)B * N * 256);  // GEMM ping (67 MB)
  float* XC   = allocF((size_t)B * N * 256);  // GEMM pong (67 MB)

  // ---- layer table (leaf offset into params flattened jax-tree order) ----
  struct LDesc { int leaf, cin, cout; };
  const LDesc L[21] = {
    {38,  67,  64}, {44,  64,  64}, {50,  64, 128},   // sc1   0-2
    { 8, 259, 128}, {14, 128, 128}, {20, 128, 128},   // fe    3-5
    {56, 131, 128}, {62, 128, 128}, {68, 128, 256},   // sc2   6-8
    {74, 259, 256}, {80, 256, 256}, {86, 256, 512},   // sc3   9-11
    {92, 771, 256}, {98, 256, 256},                   // up1b  12-13
    {104, 259, 128}, {110, 128, 128}, {116, 128, 256},// up2a  14-16
    {122, 512, 256},                                  // up2b  17
    {26, 320, 256}, {32, 256, 256},                   // fp    18-19
    { 0, 256, 128},                                   // cls1  20
  };
  _Float16* w16[21]; float* wb[21]; int wkp[21];
  for (int i = 0; i < 21; ++i) {
    int kp = (L[i].cin + 31) & ~31;
    wkp[i] = kp;
    w16[i] = allocH((size_t)L[i].cout * kp);
    wb[i]  = allocF((size_t)L[i].cout);
    int tot = L[i].cout * kp;
    k_prep_w<<<(tot + 255) / 256, 256, 0, stream>>>(
        pl(L[i].leaf + 0), pl(L[i].leaf + 1), pl(L[i].leaf + 2),
        pl(L[i].leaf + 3), pl(L[i].leaf + 4), pl(L[i].leaf + 5),
        L[i].cin, kp, L[i].cout, w16[i], wb[i]);
  }

  auto gemm = [&](const float* X, float* Y, int R, int li) {
    dim3 g((unsigned)((R + 127) / 128), (unsigned)((L[li].cout + 63) / 64));
    k_gemm<<<g, 256, 0, stream>>>(X, w16[li], wb[li], Y, R, wkp[li], L[li].cout, 1);
  };
  auto chain = [&](const float* Xin, int R, int first, int cnt) -> const float* {
    const float* cur = Xin;
    for (int i = 0; i < cnt; ++i) {
      float* out = (i & 1) ? XC : XB;
      gemm(cur, out, R, first + i);
      cur = out;
    }
    return cur;
  };

  // ---- transpose features ----
  {
    int tot = B * 64 * N;
    k_transpose<<<(tot + 255) / 256, 256, 0, stream>>>(features1, f1t, B, 64, N);
    k_transpose<<<(tot + 255) / 256, 256, 0, stream>>>(features2, f2t, B, 64, N);
  }

  // ---- set_conv1 on both clouds ----
  for (int cloud = 0; cloud < 2; ++cloud) {
    const float* pts = cloud ? points2 : points1;
    const float* ft  = cloud ? f2t : f1t;
    float* nxyz = cloud ? p2_2 : p1_2;
    float* fout = cloud ? f2_2 : f1_2;
    k_fps<<<B, 256, 0, stream>>>(pts, N, 256, fpsi, nxyz);
    k_group<<<B * 256, 64, 0, stream>>>(pts, ft, nxyz, N, 64, 256, 16, 96, 1.0f, XA);
    const float* y = chain(XA, B * 256 * 16, 0, 3);
    int tot = B * 256 * 128;
    k_maxpool<<<(tot + 255) / 256, 256, 0, stream>>>(y, fout, B * 256, 16, 128);
  }

  // ---- flow embedding (chunked per batch) ----
  k_knn<<<(B * 256 + 255) / 256, 256, 0, stream>>>(p1_2, p2_2, B, 256, 256, 64, fi, (float*)nullptr);
  for (int b = 0; b < B; ++b) {
    int tot = 256 * 64 * 288;
    k_group_fe<<<(tot + 255) / 256, 256, 0, stream>>>(
        f1_2 + (size_t)b * 256 * 128, f2_2 + (size_t)b * 256 * 128,
        p1_2 + (size_t)b * 256 * 3,   p2_2 + (size_t)b * 256 * 3,
        fi + (size_t)b * 256 * 64, 256, 64, 288, XA);
    const float* y = chain(XA, 256 * 64, 3, 3);
    int t2 = 256 * 128;
    k_maxpool<<<(t2 + 255) / 256, 256, 0, stream>>>(y, emb + (size_t)b * 256 * 128, 256, 64, 128);
  }

  // ---- set_conv2 ----
  k_fps<<<B, 256, 0, stream>>>(p1_2, 256, 64, fpsi, p1_3);
  k_group<<<B * 64, 64, 0, stream>>>(p1_2, emb, p1_3, 256, 128, 64, 8, 160, 4.0f, XA);
  {
    const float* y = chain(XA, B * 64 * 8, 6, 3);
    int tot = B * 64 * 256;
    k_maxpool<<<(tot + 255) / 256, 256, 0, stream>>>(y, f1_3, B * 64, 8, 256);
  }

  // ---- set_conv3 ----
  k_fps<<<B, 256, 0, stream>>>(p1_3, 64, 16, fpsi, p1_4);
  k_group<<<B * 16, 64, 0, stream>>>(p1_3, f1_3, p1_4, 64, 256, 16, 8, 288, 16.0f, XA);
  {
    const float* y = chain(XA, B * 16 * 8, 9, 3);
    int tot = B * 16 * 512;
    k_maxpool<<<(tot + 255) / 256, 256, 0, stream>>>(y, f1_4, B * 16, 8, 512);
  }

  // ---- set_upconv1 ----
  k_knn<<<(B * 64 + 255) / 256, 256, 0, stream>>>(p1_3, p1_4, B, 64, 16, 8, ui1, (float*)nullptr);
  {
    int tot = B * 64 * 800;
    k_up1_build<<<(tot + 255) / 256, 256, 0, stream>>>(f1_4, p1_4, p1_3, f1_3, ui1, B, XA);
  }
  gemm(XA, XB, B * 64, 12);
  gemm(XB, nf13, B * 64, 13);

  // ---- set_upconv2 ----
  k_knn<<<(B * 256 + 255) / 256, 256, 0, stream>>>(p1_2, p1_3, B, 256, 64, 8, ui2, (float*)nullptr);
  {
    int tot = B * 256 * 8 * 288;
    k_group_up2<<<(tot + 255) / 256, 256, 0, stream>>>(nf13, p1_3, p1_2, ui2, B, XA);
  }
  {
    const float* y = chain(XA, B * 256 * 8, 14, 3);   // ends in XB
    int tot = B * 256 * 256;
    k_maxpool<<<(tot + 255) / 256, 256, 0, stream>>>(y, XC, B * 256, 8, 256);
    int tot2 = B * 256 * 512;
    k_cat_up2<<<(tot2 + 255) / 256, 256, 0, stream>>>(XC, f1_2, emb, B, XA);
    gemm(XA, nf12, B * 256, 17);
  }

  // ---- feature propagation + classifier ----
  k_knn<<<(B * N + 255) / 256, 256, 0, stream>>>(points1, p1_2, B, N, 256, 3, pi, pd2);
  {
    int tot = B * N * 320;
    k_fp_build<<<(tot + 255) / 256, 256, 0, stream>>>(nf12, f1t, pi, pd2, B, N, XA);
  }
  {
    const float* y = chain(XA, B * N, 18, 2);         // XB -> XC
    gemm(y, XB, B * N, 20);                           // cls1 -> [B*N,128]
    k_cls2<<<(B * N + 255) / 256, 256, 0, stream>>>(XB, pl(6), pl(7), B, N, (float*)d_out);
  }
}
